// MessagePassingConvolution_54382875902593
// MI455X (gfx1250) — matrix operand order repaired and checked by reading, not compiled, not run on hardware
//
#include <hip/hip_runtime.h>
#include <hip/hip_bf16.h>

typedef __attribute__((ext_vector_type(16))) _Float16 v16h;
typedef __attribute__((ext_vector_type(8)))  _Float16 v8h;
typedef __attribute__((ext_vector_type(8)))  float    v8f;

#define TILE_E     128
#define THREADS    256
#define INV_SQRT3f 0.57735026918962576f
#define RSQRT8f    0.35355339059327373f             // 1/sqrt(8)  -> folded into W0
#define L2SCALEf   0.125f                           // 1/sqrt(64) -> folded into W1
#define L3SCALEf   (0.125f * 0.40824829046386302f)  // 1/8 * 1/sqrt(6) -> folded into W2

// Packed f16 B-fragment image in workspace (halfword offsets):
//   [    0,  2048)  W0: 4 tiles  (K padded 8->32)
//   [ 2048,  6144)  W1: 8 tiles  (nt,kc)
//   [ 6144, 18432)  W2: 24 tiles (nt 0..11, kc 0..1); nt 0..5 = cols 0:96, nt 6..11 = cols 96:192
#define WPAK_TOTAL 18432

#define WMMA_F16(a, b, c) \
  __builtin_amdgcn_wmma_f32_16x16x32_f16(false, (a), false, (b), (short)0, (c), false, false)

// ---- A fragment: 16x32 f16 tile from row-major LDS [rows][ld] --------------
// lane L: m = L%16; lanes 0-15 hold K 0-7 & 16-23; lanes 16-31 hold K 8-15 & 24-31.
__device__ __forceinline__ v16h lda(const _Float16* __restrict__ src, int ld, int row0, int k0) {
  int lane = threadIdx.x & 31;
  int hf = lane >> 4, m = lane & 15;
  const _Float16* p = src + (row0 + m) * ld + k0 + hf * 8;
  v8h lo = *(const v8h*)(p);        // K = base + 0..7
  v8h hi = *(const v8h*)(p + 16);   // K = base + 16..23
  v16h a;
#pragma unroll
  for (int j = 0; j < 8; ++j) { a[j] = lo[j]; a[8 + j] = hi[j]; }
  return a;
}

// ---- B fragment from pre-swizzled weights: tile*512 + lane*16 halves -------
__device__ __forceinline__ v16h ldbp(const _Float16* __restrict__ base, int tile) {
  return *(const v16h*)(base + (tile << 9) + (threadIdx.x & 31) * 16);
}

// ---- Bulk LDS copy, 16B granules (global b128 -> ds b128) ------------------
__device__ __forceinline__ void lds_copy16(_Float16* __restrict__ dst,
                                           const _Float16* __restrict__ src, int halves) {
  uint4* d = (uint4*)dst;
  const uint4* s = (const uint4*)src;
  for (int i = threadIdx.x; i < (halves >> 3); i += THREADS) d[i] = s[i];
}

// ---- C/D 16x16 f32 store: lane L: n = L%16; VGPR r holds M = r + 8*(L/16) --
__device__ __forceinline__ void st_swish(_Float16* __restrict__ dst, int ld, int row0, int col0, v8f c) {
  int lane = threadIdx.x & 31;
  int hf = lane >> 4, n = lane & 15;
#pragma unroll
  for (int r = 0; r < 8; ++r) {
    float x = c[r];
    float y = x * __builtin_amdgcn_rcpf(1.0f + __expf(-x));  // swish, fast rcp
    dst[(row0 + 8 * hf + r) * ld + col0 + n] = (_Float16)y;
  }
}

__device__ __forceinline__ void st_raw(_Float16* __restrict__ dst, int ld, int row0, int col0, v8f c) {
  int lane = threadIdx.x & 31;
  int hf = lane >> 4, n = lane & 15;
#pragma unroll
  for (int r = 0; r < 8; ++r)
    dst[(row0 + 8 * hf + r) * ld + col0 + n] = (_Float16)c[r];
}

// ---------------------------------------------------------------------------

__global__ void zero_out_kernel(float* __restrict__ p, size_t n) {
  size_t i = (size_t)blockIdx.x * blockDim.x + threadIdx.x;
  size_t stride = (size_t)gridDim.x * blockDim.x;
  for (; i < n; i += stride) p[i] = 0.0f;
}

// One-time per launch: build packed f16 B-fragment weight image in workspace.
// Packed idx within a matrix: ((tile*32 + lane)*16 + slot), tile = nt*nKC + kc;
// lane = hf*16 + ncol; source K = kc*32 + hf*8 + (slot&7) + (slot>>3)*16, N = nt*16 + ncol.
__global__ void pack_weights_kernel(const float* __restrict__ W0,
                                    const float* __restrict__ W1,
                                    const float* __restrict__ W2,
                                    _Float16* __restrict__ wpak) {
  int idx = blockIdx.x * blockDim.x + threadIdx.x;
  int gstride = gridDim.x * blockDim.x;
  for (; idx < WPAK_TOTAL; idx += gstride) {
    const float* src; int p, nKC, strideN, Kreal; float scale;
    if (idx < 2048)      { src = W0; p = idx;        nKC = 1; strideN = 64;  Kreal = 8;  scale = RSQRT8f; }
    else if (idx < 6144) { src = W1; p = idx - 2048; nKC = 2; strideN = 64;  Kreal = 64; scale = L2SCALEf; }
    else                 { src = W2; p = idx - 6144; nKC = 2; strideN = 192; Kreal = 64; scale = L3SCALEf; }
    int slot = p & 15;
    int lane = (p >> 4) & 31;
    int tile = p >> 9;
    int kc = tile % nKC, nt = tile / nKC;
    int hf = lane >> 4, ncol = lane & 15;
    int k = kc * 32 + hf * 8 + (slot & 7) + ((slot >> 3) << 4);
    int n = nt * 16 + ncol;
    float v = (k < Kreal) ? src[k * strideN + n] * scale : 0.0f;
    wpak[idx] = (_Float16)v;
  }
}

__global__ __launch_bounds__(THREADS)
void mp_conv_kernel(const float* __restrict__ node_s,    // [N,32]
                    const float* __restrict__ node_v,    // [N,32,3]
                    const float* __restrict__ edge_s0,   // [E,1]
                    const float* __restrict__ edge_sh1,  // [E,3]
                    const float* __restrict__ radial,    // [E,8]
                    const _Float16* __restrict__ wpak,   // packed weight image (ws)
                    const int*   __restrict__ senders,
                    const int*   __restrict__ receivers,
                    float* __restrict__ out,             // [N,384]
                    int E) {
  // Exactly 64 KB LDS. W2 is staged per 96-col chunk; sWt aliases dead sA+sH1.
  __shared__ __align__(64) char smem[65536];
  _Float16* sW0 = (_Float16*)(smem);            // packed B-frags: 4 tiles    4 KB
  _Float16* sW1 = (_Float16*)(smem + 4096);     // packed B-frags: 8 tiles    8 KB
  _Float16* sW2 = (_Float16*)(smem + 12288);    // packed B-frags: 12 tiles  12 KB (one 96-col chunk)
  _Float16* sA  = (_Float16*)(smem + 24576);    // 128x32 radial (row-major)  8 KB
  _Float16* sH1 = (_Float16*)(smem + 32768);    // 128x64 hidden1            16 KB
  _Float16* sWt = (_Float16*)(smem + 24576);    // 128x96 w-chunk (alias)    24 KB
  _Float16* sH2 = (_Float16*)(smem + 49152);    // 128x64 hidden2            16 KB

  const int tid  = threadIdx.x;
  const int wave = tid >> 5;                    // == M-tile row owned by this wave
  const int e0   = blockIdx.x * TILE_E;

  // ---- Edge / node gather (prefetch; overlaps staging + WMMA) ----
  const int  eloc  = tid >> 1;      // 2 threads per edge
  const int  q     = tid & 1;       // channel half: [16q, 16q+16)
  const int  e     = e0 + eloc;
  const bool valid = (e < E);

  int   rcv = 0;
  float s0 = 0.f, sh0 = 0.f, sh1 = 0.f, sh2 = 0.f;
  float ms[16];
  float mv[48];
#pragma unroll
  for (int j = 0; j < 16; ++j) ms[j] = 0.f;
#pragma unroll
  for (int j = 0; j < 48; ++j) mv[j] = 0.f;

  if (valid) {
    int snd = senders[e];
    rcv = receivers[e];
    s0  = edge_s0[e];
    sh0 = edge_sh1[3 * e + 0];
    sh1 = edge_sh1[3 * e + 1];
    sh2 = edge_sh1[3 * e + 2];
    const float4* ps = reinterpret_cast<const float4*>(node_s + (size_t)snd * 32 + q * 16);
#pragma unroll
    for (int j = 0; j < 4; ++j) {
      float4 t = ps[j];
      ms[4 * j + 0] = t.x; ms[4 * j + 1] = t.y; ms[4 * j + 2] = t.z; ms[4 * j + 3] = t.w;
    }
    const float4* pv = reinterpret_cast<const float4*>(node_v + (size_t)snd * 96 + q * 48);
#pragma unroll
    for (int j = 0; j < 12; ++j) {
      float4 t = pv[j];
      mv[4 * j + 0] = t.x; mv[4 * j + 1] = t.y; mv[4 * j + 2] = t.z; mv[4 * j + 3] = t.w;
    }
  }

  // ---- Stage weights: W0|W1|W2-chunk0 is one contiguous 24 KB copy ----
  lds_copy16(sW0, wpak, 12288);

  // ---- Stage radial [128x8] -> row-major [128x32] f16, zero-padded, u32 packed ----
  {
    unsigned* sAw = (unsigned*)sA;
    for (int i = tid; i < TILE_E * 16; i += THREADS) {
      int r = i >> 4, kp = i & 15;
      int k = kp * 2;
      int er = e0 + r;
      float f0 = (k < 8 && er < E) ? radial[(size_t)er * 8 + k] : 0.0f;
      float f1 = (k + 1 < 8 && er < E) ? radial[(size_t)er * 8 + k + 1] : 0.0f;
      union { _Float16 h[2]; unsigned u; } pk;
      pk.h[0] = (_Float16)f0; pk.h[1] = (_Float16)f1;
      sAw[i] = pk.u;
    }
  }
  __syncthreads();

  // ---- Layer 1: [128x32] @ [32x64] -> swish -> sH1. Wave owns M-tile row = wave ----
  {
    v16h a = lda(sA, 32, wave * 16, 0);
#pragma unroll
    for (int nt = 0; nt < 4; ++nt) {
      v8f c = {};
      c = WMMA_F16(a, ldbp(sW0, nt), c);
      st_swish(sH1, 64, wave * 16, nt * 16, c);
    }
  }
  __syncthreads();

  // ---- Layer 2: [128x64] @ [64x64] -> swish -> sH2 (A fragments hoisted) ----
  {
    v16h a0 = lda(sH1, 64, wave * 16, 0);
    v16h a1 = lda(sH1, 64, wave * 16, 32);
#pragma unroll
    for (int nt = 0; nt < 4; ++nt) {
      v8f c = {};
      c = WMMA_F16(a0, ldbp(sW1, nt * 2 + 0), c);
      c = WMMA_F16(a1, ldbp(sW1, nt * 2 + 1), c);
      st_swish(sH2, 64, wave * 16, nt * 16, c);
    }
  }
  __syncthreads();  // also fences sA/sH1 before sWt aliases them

  // ================= Chunk 0: w cols [0,96) -> scalar-irrep outputs =================
  {
    v16h a0 = lda(sH2, 64, wave * 16, 0);
    v16h a1 = lda(sH2, 64, wave * 16, 32);
#pragma unroll
    for (int nt = 0; nt < 6; ++nt) {
      v8f c = {};
      c = WMMA_F16(a0, ldbp(sW2, nt * 2 + 0), c);
      c = WMMA_F16(a1, ldbp(sW2, nt * 2 + 1), c);
      st_raw(sWt, 96, wave * 16, nt * 16, c);
    }
  }
  __syncthreads();

  if (valid) {
    float* orow = out + (size_t)rcv * 384;
    const _Float16* wb = sWt + eloc * 96 + q * 16;
    v16h w0 = *(const v16h*)(wb);        // w[c]    (cols 0:32  -> msg_s)
    v16h w1 = *(const v16h*)(wb + 32);   // w[32+c] (cols 32:64 -> tp_ss)
    v16h w2 = *(const v16h*)(wb + 64);   // w[64+c] (cols 64:96 -> tp_vv)
#pragma unroll
    for (int j = 0; j < 16; ++j) {
      int c = q * 16 + j;
      float s = ms[j];
      float dotc = (mv[3 * j] * sh0 + mv[3 * j + 1] * sh1 + mv[3 * j + 2] * sh2) * INV_SQRT3f;
      atomicAdd(orow + c,       s * (float)w0[j]);
      atomicAdd(orow + 32 + c,  s * s0 * (float)w1[j]);
      atomicAdd(orow + 64 + c,  dotc * (float)w2[j]);
    }
  }
  // Restage W2 cols [96,192): second contiguous 12 KB of the packed image.
  lds_copy16(sW2, wpak + 12288, 6144);
  __syncthreads();

  // ================= Chunk 1: w cols [96,192) -> vector-irrep outputs =================
  {
    v16h a0 = lda(sH2, 64, wave * 16, 0);
    v16h a1 = lda(sH2, 64, wave * 16, 32);
#pragma unroll
    for (int nt = 0; nt < 6; ++nt) {
      v8f c = {};
      c = WMMA_F16(a0, ldbp(sW2, nt * 2 + 0), c);
      c = WMMA_F16(a1, ldbp(sW2, nt * 2 + 1), c);
      st_raw(sWt, 96, wave * 16, nt * 16, c);
    }
  }
  __syncthreads();

  if (valid) {
    float* orow = out + (size_t)rcv * 384 + 96;
    const _Float16* wb = sWt + eloc * 96 + q * 16;
    v16h w0 = *(const v16h*)(wb);        // cols 96:128  -> msg_v weight
    v16h w1 = *(const v16h*)(wb + 32);   // cols 128:160 -> tp_sv weight
    v16h w2 = *(const v16h*)(wb + 64);   // cols 160:192 -> tp_vs weight
#pragma unroll
    for (int j = 0; j < 16; ++j) {
      int c = q * 16 + j;
      float s  = ms[j];
      float v0 = mv[3 * j], v1 = mv[3 * j + 1], v2 = mv[3 * j + 2];
      float w_mv = (float)w0[j];
      float w_sv = (float)w1[j];
      float w_vs = (float)w2[j] * s0;
      atomicAdd(orow + 3 * c + 0, v0 * w_mv);
      atomicAdd(orow + 3 * c + 1, v1 * w_mv);
      atomicAdd(orow + 3 * c + 2, v2 * w_mv);
      atomicAdd(orow + 3 * (32 + c) + 0, s * sh0 * w_sv);
      atomicAdd(orow + 3 * (32 + c) + 1, s * sh1 * w_sv);
      atomicAdd(orow + 3 * (32 + c) + 2, s * sh2 * w_sv);
      atomicAdd(orow + 3 * (64 + c) + 0, v0 * w_vs);
      atomicAdd(orow + 3 * (64 + c) + 1, v1 * w_vs);
      atomicAdd(orow + 3 * (64 + c) + 2, v2 * w_vs);
    }
  }
}

extern "C" void kernel_launch(void* const* d_in, const int* in_sizes, int n_in,
                              void* d_out, int out_size, void* d_ws, size_t ws_size,
                              hipStream_t stream) {
  (void)n_in; (void)ws_size;
  const float* node_s    = (const float*)d_in[0];
  const float* node_v    = (const float*)d_in[1];
  const float* edge_s0   = (const float*)d_in[2];
  const float* edge_sh1  = (const float*)d_in[3];
  const float* radial    = (const float*)d_in[4];
  const float* W0        = (const float*)d_in[5];
  const float* W1        = (const float*)d_in[6];
  const float* W2        = (const float*)d_in[7];
  const int*   senders   = (const int*)d_in[8];
  const int*   receivers = (const int*)d_in[9];
  float*       out       = (float*)d_out;
  _Float16*    wpak      = (_Float16*)d_ws;   // 36 KB packed weight image

  const int E = in_sizes[8];  // number of edges

  // d_out is accumulated with atomics and poisoned by the harness: zero it each call.
  zero_out_kernel<<<2048, THREADS, 0, stream>>>(out, (size_t)out_size);
  // Build the packed f16 B-fragment weight image once per launch.
  pack_weights_kernel<<<72, THREADS, 0, stream>>>(W0, W1, W2, wpak);

  const int blocks = (E + TILE_E - 1) / TILE_E;
  mp_conv_kernel<<<blocks, THREADS, 0, stream>>>(node_s, node_v, edge_s0, edge_sh1, radial,
                                                 wpak, senders, receivers, out, E);
}
